// HeteroGNN_23321672417703
// MI455X (gfx1250) — matrix operand order repaired and checked by reading, not compiled, not run on hardware
//
#include <hip/hip_runtime.h>
#include <hip/hip_bf16.h>

// ---------------------------------------------------------------------------
// Types for CDNA5 WMMA (wave32)
// ---------------------------------------------------------------------------
typedef __bf16 bf16_t;
typedef __attribute__((ext_vector_type(16))) __bf16 v16bf;
typedef __attribute__((ext_vector_type(8)))  float  v8f;
typedef __attribute__((ext_vector_type(4)))  int    v4i_t;

union Frag16 {
    v16bf v;
    uint4 u[2];   // 32 bytes = 16 bf16
};

// gfx1250 async global->LDS path (ASYNCcnt-tracked), with safe fallback.
#if defined(__has_builtin)
#  if __has_builtin(__builtin_amdgcn_global_load_async_to_lds_b128)
#    define HAVE_ASYNC_LDS 1
#  else
#    define HAVE_ASYNC_LDS 0
#  endif
#  if __has_builtin(__builtin_amdgcn_s_wait_asynccnt)
#    define HAVE_WAIT_ASYNC 1
#  else
#    define HAVE_WAIT_ASYNC 0
#  endif
#else
#  define HAVE_ASYNC_LDS 0
#  define HAVE_WAIT_ASYNC 0
#endif

__device__ __forceinline__ bf16_t f2bf(float f) {
    union { float f; unsigned u; } a; a.f = f;
    unsigned u = a.u;
    u += 0x7FFFu + ((u >> 16) & 1u);      // round-to-nearest-even
    unsigned short hv = (unsigned short)(u >> 16);
    bf16_t r;
    __builtin_memcpy(&r, &hv, 2);
    return r;
}

// ---------------------------------------------------------------------------
// Elementwise / utility kernels
// ---------------------------------------------------------------------------
__global__ void zero_f32_kernel(float* __restrict__ p, int n) {
    int t = blockIdx.x * blockDim.x + threadIdx.x;
    if (t < n) p[t] = 0.0f;
}

__global__ void cast_bf16_kernel(const float* __restrict__ in, bf16_t* __restrict__ out, int n) {
    int t = blockIdx.x * blockDim.x + threadIdx.x;
    if (t < n) out[t] = f2bf(in[t]);
}

__global__ void count_kernel(const int* __restrict__ dst, int E, float* __restrict__ cnt) {
    int t = blockIdx.x * blockDim.x + threadIdx.x;
    if (t < E) atomicAdd(&cnt[dst[t]], 1.0f);
}

// dinv = rsqrt(indeg + 1)   (self-loop contributes +1, so always > 0)
__global__ void finalize_dinv_kernel(float* __restrict__ d, int n) {
    int t = blockIdx.x * blockDim.x + threadIdx.x;
    if (t < n) d[t] = rsqrtf(d[t] + 1.0f);
}

// winv = 1 / max(cnt, 1)
__global__ void finalize_winv_kernel(float* __restrict__ w, int n) {
    int t = blockIdx.x * blockDim.x + threadIdx.x;
    if (t < n) w[t] = 1.0f / fmaxf(w[t], 1.0f);
}

// mean aggregate of author features (D=128) over writes edges: agg[dst] += xa[src]
__global__ void scatter_mean_kernel(const float* __restrict__ xa,
                                    const int* __restrict__ src,
                                    const int* __restrict__ dst,
                                    float* __restrict__ agg, int total) {
    int t = blockIdx.x * blockDim.x + threadIdx.x;
    if (t >= total) return;
    int e = t >> 5;          // 32 float4 chunks per 128-wide row
    int c = t & 31;
    int s = src[e], d = dst[e];
    float4 v = *(((const float4*)(xa + (size_t)s * 128)) + c);
    float* o = agg + (size_t)d * 128 + (size_t)c * 4;
    atomicAdd(o + 0, v.x); atomicAdd(o + 1, v.y);
    atomicAdd(o + 2, v.z); atomicAdd(o + 3, v.w);
}

// mean_bf16 = bf16(agg * winv[row]),  D=128
__global__ void mean_finalize_kernel(const float* __restrict__ agg,
                                     const float* __restrict__ winv,
                                     bf16_t* __restrict__ mean_bf, int total) {
    int t = blockIdx.x * blockDim.x + threadIdx.x;
    if (t >= total) return;
    int i = t >> 7;
    mean_bf[t] = f2bf(agg[t] * winv[i]);
}

// GCN scatter (D=256): out[dst] += h[src] * dinv[src] * dinv[dst]
__global__ void scatter_gcn_kernel(const float* __restrict__ h,
                                   const int* __restrict__ src,
                                   const int* __restrict__ dst,
                                   const float* __restrict__ dinv,
                                   float* __restrict__ out, int total) {
    int t = blockIdx.x * blockDim.x + threadIdx.x;
    if (t >= total) return;
    int e = t >> 6;          // 64 float4 chunks per 256-wide row
    int c = t & 63;
    int s = src[e], d = dst[e];
    float nrm = dinv[s] * dinv[d];
    float4 v = *(((const float4*)(h + (size_t)s * 256)) + c);
    float* o = out + (size_t)d * 256 + (size_t)c * 4;
    atomicAdd(o + 0, v.x * nrm); atomicAdd(o + 1, v.y * nrm);
    atomicAdd(o + 2, v.z * nrm); atomicAdd(o + 3, v.w * nrm);
}

// p = relu(gcn_acc + h*dinv^2 + gcn_b + sage_sum + sage_b), cast to bf16. D=256
__global__ void combine_kernel(const float* __restrict__ gcn_acc,
                               const float* __restrict__ h,
                               const float* __restrict__ dinv,
                               const float* __restrict__ gb,
                               const float* __restrict__ sage,
                               const float* __restrict__ sb,
                               bf16_t* __restrict__ p_bf, int total) {
    int t = blockIdx.x * blockDim.x + threadIdx.x;
    if (t >= total) return;
    int i = t >> 8;
    int d = t & 255;
    float di = dinv[i];
    float v = gcn_acc[t] + h[t] * di * di + gb[d] + sage[t] + sb[d];
    p_bf[t] = f2bf(fmaxf(v, 0.0f));
}

// ---------------------------------------------------------------------------
// B-matrix pre-pack into WMMA fragment order.
// Packed element idx = (((kt*Ntiles + nt)*32 + lane)*16 + j)
//   maps to  B[kt*32 + (lane>>4)*8 + (j&7) + (j>>3)*16][nt*16 + (lane&15)]   (zero pad n>=N)
// ---------------------------------------------------------------------------
__global__ void pack_b_kernel(const float* __restrict__ B, bf16_t* __restrict__ Bp,
                              int K, int N, int Ntiles, int total) {
    int idx = blockIdx.x * blockDim.x + threadIdx.x;
    if (idx >= total) return;
    int j    = idx & 15;
    int lane = (idx >> 4) & 31;
    int rest = idx >> 9;
    int nt   = rest % Ntiles;
    int kt   = rest / Ntiles;
    int k = kt * 32 + (lane >> 4) * 8 + (j & 7) + (j >> 3) * 16;
    int n = nt * 16 + (lane & 15);
    float v = (k < K && n < N) ? B[(size_t)k * N + n] : 0.0f;
    Bp[idx] = f2bf(v);
}

// ---------------------------------------------------------------------------
// WMMA GEMM with LDS-staged B panels:
//   C[M x Nlog] (row stride ldc) = A[M x K] @ B(packed) (+bias)(+= optional)
// One wave per 16-row tile (8 waves / 256-thread block). B fragments for an
// 8-n-tile chunk are staged cooperatively into LDS (async global->LDS via the
// gfx1250 builtin when available), then each wave runs a fully unrolled
// K loop of v_wmma_f32_16x16x32_bf16 fed by ds_load_b128.
// Dynamic LDS = KTILES * 8 * 1024 bytes (64KB for K=256, 32KB for K=128).
// M must be a multiple of 16.
// ---------------------------------------------------------------------------
template <int KTILES>
__global__ __launch_bounds__(256)
void wmma_gemm_kernel(const bf16_t* __restrict__ A,
                      const bf16_t* __restrict__ Bp,
                      float* __restrict__ C,
                      int M, int Nlog, int Ntiles, long ldc,
                      const float* __restrict__ bias, int accumulate) {
    extern __shared__ char smem_raw[];
    bf16_t* Bs = (bf16_t*)smem_raw;

    const int K = KTILES * 32;
    const int NT_CHUNK = 8;

    int lane  = threadIdx.x & 31;
    int wave  = threadIdx.x >> 5;
    int mtile = blockIdx.x * 8 + wave;
    bool active = (mtile * 16 < M);       // wave-uniform
    int m0 = mtile * 16;

    // A fragments (16x32 bf16, ISA layout): lane<16 holds K {0..7,16..23} of row (lane&15)
    Frag16 afr[KTILES];
    if (active) {
        int mrow = m0 + (lane & 15);
        int kb   = (lane >> 4) * 8;
#pragma unroll
        for (int kt = 0; kt < KTILES; ++kt) {
            const bf16_t* pa = A + (size_t)mrow * K + kt * 32 + kb;
            afr[kt].u[0] = *(const uint4*)(pa);        // K = kb .. kb+7
            afr[kt].u[1] = *(const uint4*)(pa + 16);   // K = kb+16 .. kb+23
        }
    }

    int rowhi = (lane >> 4) * 8;                       // C/D: VGPR r -> row r + 8*(lane>>4)
    int nloc  = lane & 15;

    for (int ntb = 0; ntb < Ntiles; ntb += NT_CHUNK) {
        int ntc = Ntiles - ntb;
        if (ntc > NT_CHUNK) ntc = NT_CHUNK;

        // ---- cooperative stage of B fragments for this n-tile chunk ----
        // For each kt, fragments (kt, ntb..ntb+ntc) are contiguous in Bp.
        int chunks = KTILES * ntc * 64;                // 16-byte chunks (frag = 1KB)
        for (int idx = (int)threadIdx.x; idx < chunks; idx += 256) {
            int kt = idx / (ntc * 64);
            int w  = idx % (ntc * 64);
            const uint4* src = (const uint4*)(Bp + ((size_t)kt * Ntiles + ntb) * 512) + w;
            uint4*       dst = (uint4*)(Bs + (size_t)kt * NT_CHUNK * 512) + w;
#if HAVE_ASYNC_LDS
            // signature: (v4i AS1* gsrc, v4i AS3* ldst, imm offset, imm cpol)
            __builtin_amdgcn_global_load_async_to_lds_b128(
                (__attribute__((address_space(1))) v4i_t*)src,
                (__attribute__((address_space(3))) v4i_t*)dst, 0, 0);
#else
            *dst = *src;
#endif
        }
#if HAVE_ASYNC_LDS && HAVE_WAIT_ASYNC
        __builtin_amdgcn_s_wait_asynccnt(0);
#endif
        __syncthreads();

        if (active) {
            for (int ntl = 0; ntl < ntc; ++ntl) {
                v8f acc = {};
#pragma unroll
                for (int kt = 0; kt < KTILES; ++kt) {
                    Frag16 bfr;
                    const bf16_t* pb = Bs + (((size_t)kt * NT_CHUNK + ntl) * 32 + lane) * 16;
                    bfr.u[0] = *(const uint4*)(pb);
                    bfr.u[1] = *(const uint4*)(pb + 8);
                    acc = __builtin_amdgcn_wmma_f32_16x16x32_bf16(
                        false, afr[kt].v, false, bfr.v, (short)0, acc, false, false);
                }
                int col = (ntb + ntl) * 16 + nloc;
                if (col < Nlog) {
                    float badd = bias ? bias[col] : 0.0f;
#pragma unroll
                    for (int r = 0; r < 8; ++r) {
                        size_t o = (size_t)(m0 + rowhi + r) * (size_t)ldc + (size_t)col;
                        float v = acc[r] + badd;
                        C[o] = accumulate ? (C[o] + v) : v;
                    }
                }
            }
        }
        __syncthreads();
    }
}

// ---------------------------------------------------------------------------
// Host-side orchestration
// ---------------------------------------------------------------------------
static inline int ceil_div(long a, long b) { return (int)((a + b - 1) / b); }

extern "C" void kernel_launch(void* const* d_in, const int* in_sizes, int n_in,
                              void* d_out, int out_size, void* d_ws, size_t ws_size,
                              hipStream_t stream) {
    (void)in_sizes; (void)n_in; (void)out_size; (void)ws_size;

    const int NP = 100000, E = 1000000;
    const int DP = 256, DA = 128, HID = 256, OUTD = 349;
    const int NT_HID = HID / 16;        // 16
    const int NT_OUT = 22;              // ceil(349/16) -> padded 352

    // --- inputs (setup_inputs order) ---
    const float* x_paper   = (const float*)d_in[0];
    const float* x_author  = (const float*)d_in[1];
    const int*   cites     = (const int*)d_in[2];
    const int*   c_src     = cites;
    const int*   c_dst     = cites + E;
    const int*   w_src     = (const int*)d_in[3];
    const int*   w_dst     = (const int*)d_in[4];
    const float* gcn_W1    = (const float*)d_in[5];
    const float* gcn_b1    = (const float*)d_in[6];
    const float* sage_Wl1  = (const float*)d_in[7];
    const float* sage_bl1  = (const float*)d_in[8];
    const float* sage_Wr1  = (const float*)d_in[9];
    const float* gcn_W2    = (const float*)d_in[10];
    const float* gcn_b2    = (const float*)d_in[11];
    const float* sage_Wl2  = (const float*)d_in[12];
    const float* sage_bl2  = (const float*)d_in[13];
    const float* sage_Wr2  = (const float*)d_in[14];
    const float* lin_W     = (const float*)d_in[15];
    const float* lin_b     = (const float*)d_in[16];
    float*       out       = (float*)d_out;

    // --- workspace carve-up ---
    size_t off = 0;
    char* base = (char*)d_ws;
    auto carve = [&](size_t bytes) -> void* {
        void* p = base + off;
        off += (bytes + 255) & ~(size_t)255;
        return p;
    };
    float*  hbuf     = (float*)carve((size_t)NP * HID * 4);   // also reused as mean-agg (NP*128)
    float*  gcn_acc  = (float*)carve((size_t)NP * HID * 4);
    float*  sage_out = (float*)carve((size_t)NP * HID * 4);
    bf16_t* xp_bf    = (bf16_t*)carve((size_t)NP * DP * 2);
    bf16_t* mean_bf  = (bf16_t*)carve((size_t)NP * DA * 2);
    bf16_t* p_bf     = (bf16_t*)carve((size_t)NP * HID * 2);
    float*  dinv     = (float*)carve((size_t)NP * 4);
    float*  winv     = (float*)carve((size_t)NP * 4);
    bf16_t* Wp_gcn1  = (bf16_t*)carve((size_t)(DP/32)  * NT_HID * 512 * 2);
    bf16_t* Wp_wl1   = (bf16_t*)carve((size_t)(DA/32)  * NT_HID * 512 * 2);
    bf16_t* Wp_wr1   = (bf16_t*)carve((size_t)(DP/32)  * NT_HID * 512 * 2);
    bf16_t* Wp_gcn2  = (bf16_t*)carve((size_t)(HID/32) * NT_HID * 512 * 2);
    bf16_t* Wp_wl2   = (bf16_t*)carve((size_t)(DA/32)  * NT_HID * 512 * 2);
    bf16_t* Wp_wr2   = (bf16_t*)carve((size_t)(HID/32) * NT_HID * 512 * 2);
    bf16_t* Wp_lin   = (bf16_t*)carve((size_t)(HID/32) * NT_OUT * 512 * 2);

    const int TB = 256;
    auto G = [&](long n) { return dim3((unsigned)ceil_div(n, TB)); };

    // --- degrees / counts ---
    zero_f32_kernel<<<G(NP), TB, 0, stream>>>(dinv, NP);
    zero_f32_kernel<<<G(NP), TB, 0, stream>>>(winv, NP);
    count_kernel<<<G(E), TB, 0, stream>>>(c_dst, E, dinv);
    count_kernel<<<G(E), TB, 0, stream>>>(w_dst, E, winv);
    finalize_dinv_kernel<<<G(NP), TB, 0, stream>>>(dinv, NP);
    finalize_winv_kernel<<<G(NP), TB, 0, stream>>>(winv, NP);

    // --- mean-aggregate authors (shared by both layers) ---
    zero_f32_kernel<<<G((long)NP * DA), TB, 0, stream>>>(hbuf, NP * DA);
    scatter_mean_kernel<<<G((long)E * (DA / 4)), TB, 0, stream>>>(x_author, w_src, w_dst, hbuf, E * (DA / 4));
    mean_finalize_kernel<<<G((long)NP * DA), TB, 0, stream>>>(hbuf, winv, mean_bf, NP * DA);

    // --- bf16 casts / weight packing ---
    cast_bf16_kernel<<<G((long)NP * DP), TB, 0, stream>>>(x_paper, xp_bf, NP * DP);
    {
        int t1 = (DP/32)  * NT_HID * 512;
        int t2 = (DA/32)  * NT_HID * 512;
        int t3 = (HID/32) * NT_OUT * 512;
        pack_b_kernel<<<G(t1), TB, 0, stream>>>(gcn_W1,   Wp_gcn1, DP,  HID,  NT_HID, t1);
        pack_b_kernel<<<G(t2), TB, 0, stream>>>(sage_Wl1, Wp_wl1,  DA,  HID,  NT_HID, t2);
        pack_b_kernel<<<G(t1), TB, 0, stream>>>(sage_Wr1, Wp_wr1,  DP,  HID,  NT_HID, t1);
        pack_b_kernel<<<G(t1), TB, 0, stream>>>(gcn_W2,   Wp_gcn2, HID, HID,  NT_HID, t1);
        pack_b_kernel<<<G(t2), TB, 0, stream>>>(sage_Wl2, Wp_wl2,  DA,  HID,  NT_HID, t2);
        pack_b_kernel<<<G(t1), TB, 0, stream>>>(sage_Wr2, Wp_wr2,  HID, HID,  NT_HID, t1);
        pack_b_kernel<<<G(t3), TB, 0, stream>>>(lin_W,    Wp_lin,  HID, OUTD, NT_OUT, t3);
    }

    dim3 gemm_grid((unsigned)ceil_div(NP / 16, 8));
    const size_t lds8 = 8 * 8 * 512 * 2;   // 64 KB: K=256 kernels
    const size_t lds4 = 4 * 8 * 512 * 2;   // 32 KB: K=128 kernels

    // =================== layer 1 ===================
    // h1 = x_paper @ gcn_W1
    wmma_gemm_kernel<8><<<gemm_grid, TB, lds8, stream>>>(xp_bf, Wp_gcn1, hbuf, NP, HID, NT_HID, HID, nullptr, 0);
    // gcn scatter (edges only; self-loop folded into combine)
    zero_f32_kernel<<<G((long)NP * HID), TB, 0, stream>>>(gcn_acc, NP * HID);
    scatter_gcn_kernel<<<G((long)E * (HID / 4)), TB, 0, stream>>>(hbuf, c_src, c_dst, dinv, gcn_acc, E * (HID / 4));
    // sage: mean @ Wl1 then += x_paper @ Wr1
    wmma_gemm_kernel<4><<<gemm_grid, TB, lds4, stream>>>(mean_bf, Wp_wl1, sage_out, NP, HID, NT_HID, HID, nullptr, 0);
    wmma_gemm_kernel<8><<<gemm_grid, TB, lds8, stream>>>(xp_bf,   Wp_wr1, sage_out, NP, HID, NT_HID, HID, nullptr, 1);
    combine_kernel<<<G((long)NP * HID), TB, 0, stream>>>(gcn_acc, hbuf, dinv, gcn_b1, sage_out, sage_bl1, p_bf, NP * HID);

    // =================== layer 2 ===================
    wmma_gemm_kernel<8><<<gemm_grid, TB, lds8, stream>>>(p_bf, Wp_gcn2, hbuf, NP, HID, NT_HID, HID, nullptr, 0);
    zero_f32_kernel<<<G((long)NP * HID), TB, 0, stream>>>(gcn_acc, NP * HID);
    scatter_gcn_kernel<<<G((long)E * (HID / 4)), TB, 0, stream>>>(hbuf, c_src, c_dst, dinv, gcn_acc, E * (HID / 4));
    wmma_gemm_kernel<4><<<gemm_grid, TB, lds4, stream>>>(mean_bf, Wp_wl2, sage_out, NP, HID, NT_HID, HID, nullptr, 0);
    wmma_gemm_kernel<8><<<gemm_grid, TB, lds8, stream>>>(p_bf,    Wp_wr2, sage_out, NP, HID, NT_HID, HID, nullptr, 1);
    combine_kernel<<<G((long)NP * HID), TB, 0, stream>>>(gcn_acc, hbuf, dinv, gcn_b2, sage_out, sage_bl2, p_bf, NP * HID);

    // =================== output linear ===================
    wmma_gemm_kernel<8><<<gemm_grid, TB, lds8, stream>>>(p_bf, Wp_lin, out, NP, OUTD, NT_OUT, OUTD, lin_b, 0);
}
